// ActorUp_73065983639595
// MI455X (gfx1250) — compile-verified, matrix operand order checked by reading, not run on hardware
//
#include <hip/hip_runtime.h>
#include <cstdint>
#include <cstddef>

// ---- problem constants ----
#define B_TOT      32768
#define N_PTS      16
#define ZDIM       64
#define HDIM       128
#define MSG        32
#define HM_DIM     160
#define HM_PAD     192    // 6 * 32 (K padding for gate GEMM)
#define MLP_IN     1184   // 1024 + 160
#define MLP_IN_PAD 1216   // 19 * 64 (BK=64 padding, zero-filled)
#define MLP_HID    1024
#define MLP_OUT    1056   // 1024 + 32
#define N3_PAD     1152   // padded to 9 * 128

typedef __attribute__((ext_vector_type(16))) __bf16 v16bf;
typedef __attribute__((ext_vector_type(8)))  float  v8f;
typedef __attribute__((ext_vector_type(4)))  unsigned int v4u;
typedef __attribute__((ext_vector_type(8)))  int v8i;
typedef __attribute__((ext_vector_type(4)))  int v4i;

__device__ __forceinline__ v8f wmma_bf16(v16bf a, v16bf b, v8f c) {
  return __builtin_amdgcn_wmma_f32_16x16x32_bf16(false, a, false, b, (short)0, c,
                                                 false, false);
}

__device__ __forceinline__ v8f vzero8() {
  v8f z;
#pragma unroll
  for (int i = 0; i < 8; ++i) z[i] = 0.f;
  return z;
}

__device__ __forceinline__ v16bf vzero16bf() {
  union { v16bf v; uint4 q[2]; } u;
  u.q[0] = make_uint4(0u, 0u, 0u, 0u);
  u.q[1] = make_uint4(0u, 0u, 0u, 0u);
  return u.v;
}

// CDNA5 16-bit fragment layout: lanes 0-15 -> K {0..7,16..23}, lanes 16-31 ->
// K {8..15,24..31}; row/col = lane & 15. Two 16B loads from K-contiguous data.
__device__ __forceinline__ v16bf load_frag(const __bf16* base, int row,
                                           int stride, int kwin, int lane) {
  const int klo = (lane & 16) ? 8 : 0;
  const __bf16* p = base + (size_t)row * stride + kwin + klo;
  union { v16bf v; uint4 q[2]; } u;
  u.q[0] = *reinterpret_cast<const uint4*>(p);
  u.q[1] = *reinterpret_cast<const uint4*>(p + 16);
  return u.v;
}

// ---------------------------------------------------------------------------
// Tensor Data Mover: DMA a [height x width] bf16 tile (row stride
// row_stride_elems) from global memory into LDS at byte offset lds_off.
// D# per CDNA5 ISA. Tracked by TENSORcnt.
// ---------------------------------------------------------------------------
__device__ __forceinline__ void tdm_load_2d(uint32_t lds_off, const void* gptr,
                                            uint32_t width, uint32_t height,
                                            uint32_t row_stride_elems) {
  const uint64_t ga = (uint64_t)(uintptr_t)gptr;
  v4u g0;
  g0[0] = 1u;                                   // count=1 (valid), user mode
  g0[1] = lds_off;                              // lds_addr (bytes)
  g0[2] = (uint32_t)(ga & 0xffffffffu);         // global_addr[31:0]
  g0[3] = (uint32_t)((ga >> 32) & 0x01ffffffu)  // global_addr[56:32]
          | (2u << 30);                         // type=2 ("image")
  const uint32_t td0 = width;                   // tensor dims == tile dims
  const uint32_t td1 = height;
  const uint64_t s0 = row_stride_elems;         // tensor_dim0_stride (48b)
  v8i g1;
  g1[0] = (int)(1u << 16);                      // data_size=1 (2 bytes)
  g1[1] = (int)((td0 & 0xffffu) << 16);         // tensor_dim0[15:0]
  g1[2] = (int)((td0 >> 16) | ((td1 & 0xffffu) << 16));
  g1[3] = (int)((td1 >> 16) | ((width & 0xffffu) << 16));  // tile_dim0
  g1[4] = (int)(height & 0xffffu);              // tile_dim1 (tile_dim2=0)
  g1[5] = (int)(s0 & 0xffffffffu);              // dim0_stride[31:0]
  g1[6] = (int)((s0 >> 32) & 0xffffu);          // dim0_stride[47:32]
  g1[7] = 0;                                    // dim1_stride unused (2-D)
  v4i gz4 = {0, 0, 0, 0};
  v8i gz8 = {0, 0, 0, 0, 0, 0, 0, 0};
  __builtin_amdgcn_tensor_load_to_lds(g0, g1, gz4, gz4, gz8, 0);
}

// ---------------------------------------------------------------------------
// Weight transpose + cast: Wt[n][k] = bf16(W[k][n]); zero pad k>=K, n>=N.
// ---------------------------------------------------------------------------
__global__ __launch_bounds__(256) void transpose_bf16_kernel(
    const float* __restrict__ W, __bf16* __restrict__ Wt, int K, int Kpad,
    int N, int Npad) {
  size_t i = (size_t)blockIdx.x * 256 + threadIdx.x;
  if (i >= (size_t)Npad * Kpad) return;
  int n = (int)(i / Kpad);
  int k = (int)(i - (size_t)n * Kpad);
  Wt[i] = (n < N && k < K) ? (__bf16)W[(size_t)k * N + n] : (__bf16)0.f;
}

// ---------------------------------------------------------------------------
// hm kernel: per 128-row block (8 waves, 16 rows each), all WMMA:
//   h   = relu(h0 @ W_h1 + b_h1)                (K=128, 4 k-steps)
//   hm  = [h | relu(m_h)]  -> X cols 1024..1215 (incl. zero pad)
//   t   = relu(hm @ W_g1a + b_g1a)              (K padded 160->192)
//   gate= t . W_g1b + b_g1b  (shuffle reduction) -> gateg[b]
// ---------------------------------------------------------------------------
__global__ __launch_bounds__(256) void hm_kernel(
    const float* __restrict__ h0, const float* __restrict__ m_h,
    const float* __restrict__ W_h1, const float* __restrict__ b_h1,
    const float* __restrict__ W_g1a, const float* __restrict__ b_g1a,
    const float* __restrict__ W_g1b, const float* __restrict__ b_g1b,
    __bf16* __restrict__ X, float* __restrict__ gateg) {
  __shared__ __align__(16) __bf16 h0S[128][128];
  __shared__ __align__(16) __bf16 hmS[128][HM_PAD];
  __shared__ __align__(16) __bf16 Wh1t[32][128];
  __shared__ __align__(16) __bf16 Wg1at[32][HM_PAD];

  const int tid = threadIdx.x, lane = tid & 31, wid = tid >> 5;
  const int l15 = lane & 15;
  const int rhi = (lane >> 4) << 3;
  const size_t row0 = (size_t)blockIdx.x * 128;

  // stage weights (transposed, bf16, zero-padded)
  for (int i = tid; i < 32 * 128; i += 256) {
    int n = i >> 7, k = i & 127;
    Wh1t[n][k] = (__bf16)W_h1[k * 32 + n];
  }
  for (int i = tid; i < 32 * HM_PAD; i += 256) {
    int n = i / HM_PAD, k = i - n * HM_PAD;
    Wg1at[n][k] = (k < HM_DIM) ? (__bf16)W_g1a[k * 32 + n] : (__bf16)0.f;
  }
  // stage h0 block (fp32 -> bf16)
  for (int i = tid; i < 128 * 128; i += 256) {
    int r = i >> 7, c = i & 127;
    h0S[r][c] = (__bf16)h0[(row0 + r) * HDIM + c];
  }
  // stage relu(m_h) into hmS[.][32..159] and X cols 1056..1183; pads
  for (int i = tid; i < 128 * 128; i += 256) {
    int r = i >> 7, c = i & 127;
    __bf16 v = (__bf16)fmaxf(m_h[(row0 + r) * 128 + c], 0.f);
    hmS[r][32 + c] = v;
    X[(row0 + r) * MLP_IN_PAD + 1056 + c] = v;
  }
  for (int i = tid; i < 128 * 32; i += 256) {  // hmS K-pad 160..191
    int r = i >> 5, c = i & 31;
    hmS[r][HM_DIM + c] = (__bf16)0.f;
  }
  for (int i = tid; i < 128 * 32; i += 256) {  // X col-pad 1184..1215
    int r = i >> 5, c = i & 31;
    X[(row0 + r) * MLP_IN_PAD + MLP_IN + c] = (__bf16)0.f;
  }
  __syncthreads();

  // ---- h = relu(h0 @ W_h1 + b_h1), WMMA; wave handles rows wid*16..+15 ----
  v8f hacc[2] = {vzero8(), vzero8()};
#pragma unroll
  for (int ks = 0; ks < 4; ++ks) {
    v16bf a = load_frag(&h0S[0][0], wid * 16 + l15, 128, ks * 32, lane);
#pragma unroll
    for (int nt = 0; nt < 2; ++nt) {
      v16bf bb = load_frag(&Wh1t[0][0], nt * 16 + l15, 128, ks * 32, lane);
      hacc[nt] = wmma_bf16(a, bb, hacc[nt]);
    }
  }
#pragma unroll
  for (int nt = 0; nt < 2; ++nt)
#pragma unroll
    for (int v = 0; v < 8; ++v) {
      int row = wid * 16 + v + rhi, col = nt * 16 + l15;
      __bf16 hb = (__bf16)fmaxf(hacc[nt][v] + b_h1[col], 0.f);
      hmS[row][col] = hb;
      X[(row0 + row) * MLP_IN_PAD + 1024 + col] = hb;
    }
  __syncthreads();

  // ---- t = relu(hm @ W_g1a + b_g1a), WMMA over K=192 ----
  v8f tacc[2] = {vzero8(), vzero8()};
#pragma unroll
  for (int ks = 0; ks < 6; ++ks) {
    v16bf a = load_frag(&hmS[0][0], wid * 16 + l15, HM_PAD, ks * 32, lane);
#pragma unroll
    for (int nt = 0; nt < 2; ++nt) {
      v16bf bb = load_frag(&Wg1at[0][0], nt * 16 + l15, HM_PAD, ks * 32, lane);
      tacc[nt] = wmma_bf16(a, bb, tacc[nt]);
    }
  }

  // ---- gate = t . W_g1b + b_g1b (reduce 32 cols per row) ----
  const float wg_lo = W_g1b[l15];
  const float wg_hi = W_g1b[16 + l15];
  const float bga_lo = b_g1a[l15];
  const float bga_hi = b_g1a[16 + l15];
  const float bgb = b_g1b[0];
#pragma unroll
  for (int v = 0; v < 8; ++v) {
    float t0 = fmaxf(tacc[0][v] + bga_lo, 0.f);
    float t1 = fmaxf(tacc[1][v] + bga_hi, 0.f);
    float p = t0 * wg_lo + t1 * wg_hi;
    p += __shfl_xor(p, 1, 32);
    p += __shfl_xor(p, 2, 32);
    p += __shfl_xor(p, 4, 32);
    p += __shfl_xor(p, 8, 32);  // reduction stays within each 16-lane half
    if (l15 == 0) gateg[row0 + wid * 16 + v + rhi] = p + bgb;
  }
}

// ---------------------------------------------------------------------------
// Pre kernel: one wave32 per batch element (8 per block), all WMMA:
//   Z = Z0@W_Z1; Zm = Z@W_Z2a + m_Z@W_Z2b + g*W_Z2[160]; M = Zm^T Zm; F_norm.
// ---------------------------------------------------------------------------
__global__ __launch_bounds__(256) void pre_kernel(
    const float* __restrict__ Z0, const float* __restrict__ m_Z,
    const float* __restrict__ W_Z1, const float* __restrict__ W_Z2,
    const float* __restrict__ gateg, __bf16* __restrict__ X,
    __bf16* __restrict__ Zmg, float* __restrict__ Fg) {
  __shared__ __align__(16) __bf16 WZ1t[32][64];
  __shared__ __align__(16) __bf16 WZ2at[32][32];
  __shared__ __align__(16) __bf16 WZ2bt[32][128];
  __shared__ __align__(16) __bf16 Atile[8][16][128];
  __shared__ __align__(16) __bf16 Ztmp[8][16][32];
  __shared__ __align__(16) __bf16 ZmL[8][16][32];

  const int tid = threadIdx.x, lane = tid & 31, w = tid >> 5;
  const int b = blockIdx.x * 8 + w;
  const int l15 = lane & 15;
  const int rhi = (lane >> 4) << 3;

  for (int i = tid; i < 32 * 64; i += 256) {
    int n = i >> 6, k = i & 63;
    WZ1t[n][k] = (__bf16)W_Z1[k * 32 + n];
  }
  for (int i = tid; i < 32 * 32; i += 256) {
    int n = i >> 5, k = i & 31;
    WZ2at[n][k] = (__bf16)W_Z2[k * 32 + n];
  }
  for (int i = tid; i < 32 * 128; i += 256) {
    int n = i >> 7, k = i & 127;
    WZ2bt[n][k] = (__bf16)W_Z2[(32 + k) * 32 + n];
  }

  const float gate = gateg[b];

  // ---- Z = Z0 @ W_Z1 (WMMA) ----
  for (int i = lane; i < N_PTS * ZDIM; i += 32) {
    int r = i >> 6, k = i & 63;
    Atile[w][r][k] = (__bf16)Z0[(size_t)b * (N_PTS * ZDIM) + i];
  }
  __syncthreads();

  v8f zacc[2] = {vzero8(), vzero8()};
#pragma unroll
  for (int ks = 0; ks < 2; ++ks) {
    v16bf a = load_frag(&Atile[w][0][0], l15, 128, ks * 32, lane);
#pragma unroll
    for (int nt = 0; nt < 2; ++nt) {
      v16bf bb = load_frag(&WZ1t[0][0], nt * 16 + l15, 64, ks * 32, lane);
      zacc[nt] = wmma_bf16(a, bb, zacc[nt]);
    }
  }
#pragma unroll
  for (int nt = 0; nt < 2; ++nt)
#pragma unroll
    for (int v = 0; v < 8; ++v)
      Ztmp[w][v + rhi][nt * 16 + l15] = (__bf16)zacc[nt][v];
  __syncthreads();

  // ---- Zm ----
  v8f zm[2] = {vzero8(), vzero8()};
  {
    v16bf a = load_frag(&Ztmp[w][0][0], l15, 32, 0, lane);
#pragma unroll
    for (int nt = 0; nt < 2; ++nt) {
      v16bf bb = load_frag(&WZ2at[0][0], nt * 16 + l15, 32, 0, lane);
      zm[nt] = wmma_bf16(a, bb, zm[nt]);
    }
  }
  for (int i = lane; i < N_PTS * 128; i += 32) {
    int r = i >> 7, k = i & 127;
    Atile[w][r][k] = (__bf16)m_Z[(size_t)b * (N_PTS * 128) + i];
  }
  __syncthreads();
#pragma unroll
  for (int ks = 0; ks < 4; ++ks) {
    v16bf a = load_frag(&Atile[w][0][0], l15, 128, ks * 32, lane);
#pragma unroll
    for (int nt = 0; nt < 2; ++nt) {
      v16bf bb = load_frag(&WZ2bt[0][0], nt * 16 + l15, 128, ks * 32, lane);
      zm[nt] = wmma_bf16(a, bb, zm[nt]);
    }
  }
  if (lane < 16) {  // row n==2 lives in VGPR v=2 on lanes 0..15
#pragma unroll
    for (int nt = 0; nt < 2; ++nt)
      zm[nt][2] += gate * W_Z2[160 * 32 + nt * 16 + lane];
  }

#pragma unroll
  for (int nt = 0; nt < 2; ++nt)
#pragma unroll
    for (int v = 0; v < 8; ++v) {
      int r = v + rhi, c = nt * 16 + l15;
      __bf16 z = (__bf16)zm[nt][v];
      ZmL[w][r][c] = z;
      Zmg[(size_t)b * 512 + r * 32 + c] = z;
    }
  __syncthreads();

  // ---- M = Zm^T @ Zm (K padded 16->32), symmetric fragments ----
  const int klo = (lane & 16) ? 8 : 0;
  v16bf frag[2];
#pragma unroll
  for (int t = 0; t < 2; ++t) {
    v16bf f = vzero16bf();
#pragma unroll
    for (int j = 0; j < 8; ++j) f[j] = ZmL[w][klo + j][t * 16 + l15];
    frag[t] = f;
  }
  v8f macc[2][2];
#pragma unroll
  for (int tm = 0; tm < 2; ++tm)
#pragma unroll
    for (int tn = 0; tn < 2; ++tn)
      macc[tm][tn] = wmma_bf16(frag[tm], frag[tn], vzero8());

  // ---- F_norm ----
  float ss = 0.f;
#pragma unroll
  for (int tm = 0; tm < 2; ++tm)
#pragma unroll
    for (int tn = 0; tn < 2; ++tn)
#pragma unroll
      for (int v = 0; v < 8; ++v) ss += macc[tm][tn][v] * macc[tm][tn][v];
  for (int off = 16; off > 0; off >>= 1) ss += __shfl_xor(ss, off, 32);
  float Fn = sqrtf(ss) + 1.f;
  if (lane == 0) Fg[b] = Fn;

  // ---- M -> X cols 0..1023 (hm part written by hm_kernel) ----
#pragma unroll
  for (int tm = 0; tm < 2; ++tm)
#pragma unroll
    for (int tn = 0; tn < 2; ++tn)
#pragma unroll
      for (int v = 0; v < 8; ++v) {
        int r = tm * 16 + v + rhi, c = tn * 16 + l15;
        X[(size_t)b * MLP_IN_PAD + r * 32 + c] = (__bf16)macc[tm][tn][v];
      }
}

// ---------------------------------------------------------------------------
// Tiled bf16 WMMA GEMM with TDM double-buffered staging.
//   Block tile 128x128, BK=64, 8 waves (4 in M x 2 in N), 32x64 per wave.
// EPI==0: relu, bf16 store. EPI==1: scale by 1/F_norm, split Mflat / M_h.
// ---------------------------------------------------------------------------
template <int EPI>
__global__ __launch_bounds__(256) void gemm_kernel(
    const __bf16* __restrict__ A, const __bf16* __restrict__ Bt,
    const float* __restrict__ bias, int K, int Nvalid,
    __bf16* __restrict__ Yout, int ystride, float* __restrict__ MhOut,
    const float* __restrict__ Fg) {
  __shared__ __align__(16) __bf16 As[2][128][64];
  __shared__ __align__(16) __bf16 Bs[2][128][64];
  __shared__ float Fl[128];

  const int tid = threadIdx.x, lane = tid & 31, wid = tid >> 5;
  const int wm = wid & 3, wn = wid >> 2;
  const int l15 = lane & 15;
  const int rhi = (lane >> 4) << 3;
  const size_t row0 = (size_t)blockIdx.x * 128;
  const int col0 = blockIdx.y * 128;

  if (EPI == 1) {
    if (tid < 128) Fl[tid] = 1.f / Fg[row0 + tid];
  }

  v8f acc[2][4];
#pragma unroll
  for (int tm = 0; tm < 2; ++tm)
#pragma unroll
    for (int tn = 0; tn < 4; ++tn) acc[tm][tn] = vzero8();

  const uint32_t ldsA[2] = {(uint32_t)(uintptr_t)&As[0][0][0],
                            (uint32_t)(uintptr_t)&As[1][0][0]};
  const uint32_t ldsB[2] = {(uint32_t)(uintptr_t)&Bs[0][0][0],
                            (uint32_t)(uintptr_t)&Bs[1][0][0]};
  const __bf16* gA = A + row0 * (size_t)K;
  const __bf16* gB = Bt + (size_t)col0 * (size_t)K;
  const int nk = K >> 6;

  if (wid == 0) {  // prologue: DMA tile 0
    tdm_load_2d(ldsA[0], gA, 64, 128, (uint32_t)K);
    tdm_load_2d(ldsB[0], gB, 64, 128, (uint32_t)K);
  }

  for (int ks = 0; ks < nk; ++ks) {
    const int cur = ks & 1;
    if (wid == 0) {
      if (ks + 1 < nk) {  // DMA next tile into the other buffer
        tdm_load_2d(ldsA[cur ^ 1], gA + (size_t)(ks + 1) * 64, 64, 128,
                    (uint32_t)K);
        tdm_load_2d(ldsB[cur ^ 1], gB + (size_t)(ks + 1) * 64, 64, 128,
                    (uint32_t)K);
        __builtin_amdgcn_s_wait_tensorcnt(2);  // current pair complete
      } else {
        __builtin_amdgcn_s_wait_tensorcnt(0);
      }
    }
    __syncthreads();  // current tiles visible to all waves

    v16bf af[2], bf2[4];
#pragma unroll
    for (int kw = 0; kw < 2; ++kw) {
#pragma unroll
      for (int tm = 0; tm < 2; ++tm)
        af[tm] =
            load_frag(&As[cur][0][0], wm * 32 + tm * 16 + l15, 64, kw * 32, lane);
#pragma unroll
      for (int tn = 0; tn < 4; ++tn)
        bf2[tn] =
            load_frag(&Bs[cur][0][0], wn * 64 + tn * 16 + l15, 64, kw * 32, lane);
#pragma unroll
      for (int tm = 0; tm < 2; ++tm)
#pragma unroll
        for (int tn = 0; tn < 4; ++tn)
          acc[tm][tn] = wmma_bf16(af[tm], bf2[tn], acc[tm][tn]);
    }
    __syncthreads();  // all reads done before next DMA overwrites this buffer
  }

  // epilogue
#pragma unroll
  for (int tm = 0; tm < 2; ++tm)
#pragma unroll
    for (int tn = 0; tn < 4; ++tn)
#pragma unroll
      for (int v = 0; v < 8; ++v) {
        int rb = wm * 32 + tm * 16 + v + rhi;
        int cb = wn * 64 + tn * 16 + l15;
        size_t row = row0 + rb;
        int col = col0 + cb;
        float val = acc[tm][tn][v];
        if (EPI == 0) {
          val = fmaxf(val + bias[col], 0.f);
          Yout[row * (size_t)ystride + col] = (__bf16)val;
        } else {
          if (col < Nvalid) {
            val = (val + bias[col]) * Fl[rb];
            if (col < 1024)
              Yout[row * 1024 + col] = (__bf16)val;
            else
              MhOut[row * 32 + (col - 1024)] = val;
          }
        }
      }
}

// ---------------------------------------------------------------------------
// Post kernel: M_Z[b] = Zm[b] (16x32) @ M_Zflat[b] (32x32), fp32 out.
// ---------------------------------------------------------------------------
__global__ __launch_bounds__(256) void post_kernel(
    const __bf16* __restrict__ Zmg, const __bf16* __restrict__ Mf,
    float* __restrict__ MZout) {
  __shared__ __align__(16) __bf16 MtL[8][32][32];
  const int tid = threadIdx.x, lane = tid & 31, w = tid >> 5;
  const int b = blockIdx.x * 8 + w;
  const int l15 = lane & 15;
  const int rhi = (lane >> 4) << 3;

  for (int i = lane; i < 1024; i += 32) {
    int j = i >> 5, k = i & 31;
    MtL[w][k][j] = Mf[(size_t)b * 1024 + i];
  }
  __syncthreads();

  v16bf a = load_frag(Zmg + (size_t)b * 512, l15, 32, 0, lane);
  v8f acc[2];
#pragma unroll
  for (int nt = 0; nt < 2; ++nt) {
    v16bf bb = load_frag(&MtL[w][0][0], nt * 16 + l15, 32, 0, lane);
    acc[nt] = wmma_bf16(a, bb, vzero8());
  }
#pragma unroll
  for (int nt = 0; nt < 2; ++nt)
#pragma unroll
    for (int v = 0; v < 8; ++v)
      MZout[(size_t)b * 512 + (v + rhi) * 32 + nt * 16 + l15] = acc[nt][v];
}

// ---------------------------------------------------------------------------
extern "C" void kernel_launch(void* const* d_in, const int* in_sizes, int n_in,
                              void* d_out, int out_size, void* d_ws,
                              size_t ws_size, hipStream_t stream) {
  (void)in_sizes; (void)n_in; (void)out_size; (void)ws_size;
  const float* Z0    = (const float*)d_in[0];
  const float* h0    = (const float*)d_in[1];
  const float* m_Z   = (const float*)d_in[2];
  const float* m_h   = (const float*)d_in[3];
  const float* W_Z1  = (const float*)d_in[4];
  const float* W_h1  = (const float*)d_in[5];
  const float* b_h1  = (const float*)d_in[6];
  const float* W_g1a = (const float*)d_in[7];
  const float* b_g1a = (const float*)d_in[8];
  const float* W_g1b = (const float*)d_in[9];
  const float* b_g1b = (const float*)d_in[10];
  const float* W_Z2  = (const float*)d_in[11];
  const float* W1    = (const float*)d_in[12];
  const float* b1    = (const float*)d_in[13];
  const float* W2    = (const float*)d_in[14];
  const float* b2    = (const float*)d_in[15];
  const float* W3    = (const float*)d_in[16];
  const float* b3    = (const float*)d_in[17];

  char* ws = (char*)d_ws;
  size_t off = 0;
  auto carve = [&](size_t bytes) -> char* {
    char* p = ws + off;
    off += (bytes + 255) & ~(size_t)255;
    return p;
  };
  __bf16* X     = (__bf16*)carve((size_t)B_TOT * MLP_IN_PAD * 2);
  __bf16* Y1    = (__bf16*)carve((size_t)B_TOT * 1024 * 2);
  __bf16* Zmg   = (__bf16*)carve((size_t)B_TOT * 512 * 2);
  float*  Fg    = (float*)carve((size_t)B_TOT * 4);
  float*  gateg = (float*)carve((size_t)B_TOT * 4);
  __bf16* W1t   = (__bf16*)carve((size_t)1024 * MLP_IN_PAD * 2);
  __bf16* W2t   = (__bf16*)carve((size_t)1024 * 1024 * 2);
  __bf16* W3t   = (__bf16*)carve((size_t)N3_PAD * 1024 * 2);
  __bf16* Y2 = X;   // X dead after layer 1
  __bf16* Mf = Y1;  // Y1 dead after layer 2
  float* MZout = (float*)d_out;
  float* MhOut = (float*)d_out + (size_t)B_TOT * 512;

  {
    size_t n1 = (size_t)1024 * MLP_IN_PAD;
    transpose_bf16_kernel<<<(unsigned)((n1 + 255) / 256), 256, 0, stream>>>(
        W1, W1t, MLP_IN, MLP_IN_PAD, 1024, 1024);
    size_t n2 = (size_t)1024 * 1024;
    transpose_bf16_kernel<<<(unsigned)((n2 + 255) / 256), 256, 0, stream>>>(
        W2, W2t, 1024, 1024, 1024, 1024);
    size_t n3 = (size_t)N3_PAD * 1024;
    transpose_bf16_kernel<<<(unsigned)((n3 + 255) / 256), 256, 0, stream>>>(
        W3, W3t, 1024, 1024, MLP_OUT, N3_PAD);
  }

  hm_kernel<<<B_TOT / 128, 256, 0, stream>>>(h0, m_h, W_h1, b_h1, W_g1a, b_g1a,
                                             W_g1b, b_g1b, X, gateg);

  pre_kernel<<<B_TOT / 8, 256, 0, stream>>>(Z0, m_Z, W_Z1, W_Z2, gateg, X, Zmg,
                                            Fg);

  gemm_kernel<0><<<dim3(B_TOT / 128, MLP_HID / 128), 256, 0, stream>>>(
      X, W1t, b1, MLP_IN_PAD, MLP_HID, Y1, 1024, nullptr, nullptr);
  gemm_kernel<0><<<dim3(B_TOT / 128, MLP_HID / 128), 256, 0, stream>>>(
      Y1, W2t, b2, MLP_HID, MLP_HID, Y2, 1024, nullptr, nullptr);
  gemm_kernel<1><<<dim3(B_TOT / 128, N3_PAD / 128), 256, 0, stream>>>(
      Y2, W3t, b3, MLP_HID, MLP_OUT, Mf, 1024, MhOut, Fg);

  post_kernel<<<B_TOT / 8, 256, 0, stream>>>(Zmg, Mf, MZout);
}
